// RBFNet_12756052869797
// MI455X (gfx1250) — matrix operand order repaired and checked by reading, not compiled, not run on hardware
//
#include <hip/hip_runtime.h>

typedef _Float16 v16h __attribute__((ext_vector_type(16)));
typedef _Float16 h2   __attribute__((ext_vector_type(2)));
typedef float    v8f  __attribute__((ext_vector_type(8)));

namespace {
constexpr int kRows   = 262144;
constexpr int kD      = 64;
constexpr int kNC     = 256;
constexpr int kOut    = 18;
constexpr int kTiles  = kRows / 16;     // 16384 row tiles of 16
constexpr int kBlocks = 256;            // 8 waves each -> 2048 waves, 8 tiles/wave
constexpr int kWavesTotal = kBlocks * 8;

// LDS layout, indices in _Float16 units
constexpr int W1H = 0;                  // 32 frags (16 nt x 2 kc) * 512 halves
constexpr int W1L = 16384;
constexpr int W2H = 32768;              // 16 frags (2 nt2 x 8 kc2) * 512 halves
constexpr int W2L = 40960;
constexpr int FT  = 49152;              // 8 waves * 1024 halves (hi 512 | lo 512)
constexpr int SMEM_HALFS = 57344;       // 114688 bytes (<= 320KB/WGP on CDNA5)
}

__device__ __forceinline__ v16h ld_frag(const _Float16* lds, int baseH, int frag, int lane) {
  // fragment = 32 lanes x 16 halves, 32B per lane -> two ds_load_b128
  return *(const v16h*)(lds + baseH + frag * 512 + lane * 16);
}

__device__ __forceinline__ v8f wmma_f16(v16h a, v16h b, v8f c) {
  return __builtin_amdgcn_wmma_f32_16x16x32_f16(false, a, false, b, (short)0, c, false, false);
}

__global__ __launch_bounds__(256)
void rbfnet_fused(const float* __restrict__ x,  const float* __restrict__ Wr,
                  const float* __restrict__ br, const float* __restrict__ Wl,
                  const float* __restrict__ bl, float* __restrict__ out)
{
  extern __shared__ _Float16 lds[];
  const int tid = threadIdx.x;
  constexpr float SCL1 = 0.2250790790392765f;   // sqrt(2*gamma) / (2*pi): accumulate in revolutions
  constexpr float SCL2 = 0.08838834764831845f;  // sqrt(2/256) folded into W_lin

  // ---- pack W_rbf into B-fragment layout (hi/lo f16 split), scale folded ----
  for (int i = tid; i < 16384; i += 256) {
    const int f = i >> 9, r = i & 511;            // f: (nt*2+kc), r: lane*16+j
    const int lane = r >> 4, j = r & 15;
    const int K = (f & 1) * 32 + ((j >> 3) << 4) + ((lane >> 4) << 3) + (j & 7);   // B layout: K = 16*(j/8) + 8*kh + j%8
    const int n = (f >> 1) * 16 + (lane & 15);
    const float w = Wr[K * kNC + n] * SCL1;
    const _Float16 hi = (_Float16)w;
    const int off = f * 512 + lane * 16 + j;
    lds[W1H + off] = hi;
    lds[W1L + off] = (_Float16)(w - (float)hi);
  }
  // ---- pack W_lin^T (pad 18 -> 32 cols with zeros), scale folded ----
  for (int i = tid; i < 8192; i += 256) {
    const int f = i >> 9, r = i & 511;            // f: (nt2*8+kc2)
    const int lane = r >> 4, j = r & 15;
    const int K = (f & 7) * 32 + ((j >> 3) << 4) + ((lane >> 4) << 3) + (j & 7);
    const int n = (f >> 3) * 16 + (lane & 15);
    const float w = (n < kOut) ? Wl[n * kNC + K] * SCL2 : 0.0f;
    const _Float16 hi = (_Float16)w;
    const int off = f * 512 + lane * 16 + j;
    lds[W2H + off] = hi;
    lds[W2L + off] = (_Float16)(w - (float)hi);
  }
  __syncthreads();

  const int lane = tid & 31;
  const int wave = tid >> 5;
  const int m    = lane & 15;    // row (A) / column (B,C,D) within tile
  const int kh   = lane >> 4;    // lane-group half
  _Float16* feat_hi = lds + FT + wave * 1024;   // per-wave 16x32 scratch
  _Float16* feat_lo = feat_hi + 512;

  float bl2[2];
  #pragma unroll
  for (int t = 0; t < 2; ++t) {
    const int c = t * 16 + m;
    bl2[t] = (c < kOut) ? bl[c] : 0.0f;
  }

  const int gw = blockIdx.x * 8 + wave;
  #pragma unroll 1
  for (int tile = gw; tile < kTiles; tile += kWavesTotal) {
    // ---- load x tile as A fragments (hi/lo split), per ISA A-layout ----
    const float* xrow = x + (size_t)(tile * 16 + m) * kD;
    v16h xh[2], xl[2];
    #pragma unroll
    for (int kc = 0; kc < 2; ++kc) {
      #pragma unroll
      for (int v = 0; v < 8; ++v) {
        const int d = kc * 32 + (v < 4 ? kh * 8 + 2 * v : 16 + kh * 8 + 2 * (v - 4));
        const float a0 = xrow[d], a1 = xrow[d + 1];
        const _Float16 h0 = (_Float16)a0, h1 = (_Float16)a1;
        xh[kc][2 * v]     = h0;  xh[kc][2 * v + 1] = h1;
        xl[kc][2 * v]     = (_Float16)(a0 - (float)h0);
        xl[kc][2 * v + 1] = (_Float16)(a1 - (float)h1);
      }
    }

    v8f c2[2];
    #pragma unroll
    for (int t = 0; t < 2; ++t)
      #pragma unroll
      for (int r = 0; r < 8; ++r) c2[t][r] = bl2[t];

    #pragma unroll 1
    for (int ntp = 0; ntp < 8; ++ntp) {          // pairs of 16-wide feature tiles
      #pragma unroll
      for (int hlf = 0; hlf < 2; ++hlf) {
        const int nt = ntp * 2 + hlf;
        const float bval = br[nt * 16 + m];      // b_rbf folded into accumulator init
        v8f c1;
        #pragma unroll
        for (int r = 0; r < 8; ++r) c1[r] = bval;
        #pragma unroll
        for (int kc = 0; kc < 2; ++kc) {         // K = 64 in two 32-chunks
          const v16h bh = ld_frag(lds, W1H, nt * 2 + kc, lane);
          const v16h bo = ld_frag(lds, W1L, nt * 2 + kc, lane);
          c1 = wmma_f16(xh[kc], bh, c1);         // hi*hi
          c1 = wmma_f16(xh[kc], bo, c1);         // hi*lo
          c1 = wmma_f16(xl[kc], bh, c1);         // lo*hi
        }
        // cos(2*pi * rev) via v_cos_f32; split result hi/lo for GEMM2
        #pragma unroll
        for (int r = 0; r < 8; ++r) {
          const float fv = __builtin_amdgcn_cosf(c1[r]);
          const _Float16 fh = (_Float16)fv;
          const int row = r + kh * 8;            // C/D layout: M = vgpr + 8*kh
          const int col = hlf * 16 + m;
          feat_hi[row * 32 + col] = fh;
          feat_lo[row * 32 + col] = (_Float16)(fv - (float)fh);
        }
      }
      // ---- transpose C-layout feat -> A-layout via per-wave LDS scratch ----
      v16h a2h, a2l;
      #pragma unroll
      for (int v = 0; v < 8; ++v) {
        const int kk = (v < 4 ? kh * 8 + 2 * v : 16 + kh * 8 + 2 * (v - 4));
        const h2 ph = *(const h2*)(feat_hi + m * 32 + kk);
        const h2 pl = *(const h2*)(feat_lo + m * 32 + kk);
        a2h[2 * v] = ph.x;  a2h[2 * v + 1] = ph.y;
        a2l[2 * v] = pl.x;  a2l[2 * v + 1] = pl.y;
      }
      // ---- GEMM2 partial accumulate (K-chunk = this nt pair) ----
      #pragma unroll
      for (int t = 0; t < 2; ++t) {
        const v16h bh = ld_frag(lds, W2H, t * 8 + ntp, lane);
        const v16h bo = ld_frag(lds, W2L, t * 8 + ntp, lane);
        c2[t] = wmma_f16(a2h, bh, c2[t]);
        c2[t] = wmma_f16(a2h, bo, c2[t]);
        c2[t] = wmma_f16(a2l, bh, c2[t]);
      }
    }

    // ---- store the 18 valid output columns (predicated) ----
    #pragma unroll
    for (int t = 0; t < 2; ++t) {
      const int col = t * 16 + m;
      if (col < kOut) {
        #pragma unroll
        for (int r = 0; r < 8; ++r) {
          out[(size_t)(tile * 16 + r + kh * 8) * kOut + col] = c2[t][r];
        }
      }
    }
  }
}

extern "C" void kernel_launch(void* const* d_in, const int* in_sizes, int n_in,
                              void* d_out, int out_size, void* d_ws, size_t ws_size,
                              hipStream_t stream) {
  (void)in_sizes; (void)n_in; (void)d_ws; (void)ws_size; (void)out_size;
  const float* x  = (const float*)d_in[0];
  const float* Wr = (const float*)d_in[1];
  const float* br = (const float*)d_in[2];
  const float* Wl = (const float*)d_in[3];
  const float* bl = (const float*)d_in[4];
  float* out = (float*)d_out;

  static_assert(SMEM_HALFS * sizeof(_Float16) == 114688, "LDS layout");
  // >64KB dynamic LDS: raise the limit (no-op in compile-only, needed on HW).
  hipFuncSetAttribute((const void*)rbfnet_fused,
                      hipFuncAttributeMaxDynamicSharedMemorySize,
                      SMEM_HALFS * (int)sizeof(_Float16));
  rbfnet_fused<<<dim3(kBlocks), dim3(256), SMEM_HALFS * sizeof(_Float16), stream>>>(
      x, Wr, br, Wl, bl, out);
}
